// RelativeMultiHeadAttn_89532888253107
// MI455X (gfx1250) — matrix operand order repaired and checked by reading, not compiled, not run on hardware
//
#include <hip/hip_runtime.h>

// ---------- types ----------
typedef __attribute__((ext_vector_type(16))) __bf16 v16bf;
typedef __attribute__((ext_vector_type(8)))  float  v8f;
typedef __attribute__((ext_vector_type(4)))  unsigned int u32x4;
typedef __attribute__((ext_vector_type(4)))  int          i32x4;
typedef __attribute__((ext_vector_type(8)))  int          i32x8;

struct __attribute__((aligned(16))) U128 { unsigned int  w[4]; };
struct __attribute__((aligned(16))) F128 { float         f[4]; };
struct __attribute__((aligned(8)))  US4  { unsigned short s[4]; };

union V16 {
    v16bf          v;
    unsigned short s[16];
    U128           q[2];
};
static_assert(sizeof(v16bf) == 32, "v16bf size");

__device__ __forceinline__ unsigned short f2bf(float f) {
    return __builtin_bit_cast(unsigned short, (__bf16)f);   // native RNE cvt
}
__device__ __forceinline__ float bf2f(unsigned short h) {
    union { unsigned u; float f; } x; x.u = ((unsigned)h) << 16;
    return x.f;
}

#define WMMA_BF16(A, B, C) \
    __builtin_amdgcn_wmma_f32_16x16x32_bf16(false, (A), false, (B), (short)0, (C), false, false)

// TDM: only the 5-arg ROCm form; compile-safe fallback otherwise (incl. host pass).
#if __has_builtin(__builtin_amdgcn_tensor_load_to_lds) && !__has_include(<hip/amd_detail/amd_gfx1250_TDM.h>)
#define USE_TDM 1
#else
#define USE_TDM 0
#endif

// Problem constants: B=4, L=1024, D=1024, NH=16, HD=64, pos rows = 2L = 2048.

// ============================================================================
// Kernel 0: bulk f32 -> bf16 conversion (x, W_qkv), 4 elements/thread.
// ============================================================================
__global__ void cvt_bf16_kernel(const float* __restrict__ src,
                                unsigned short* __restrict__ dst, int n4)
{
    int idx = blockIdx.x * 256 + threadIdx.x;
    if (idx >= n4) return;
    F128 f = *(const F128*)(src + (size_t)idx * 4);
    US4 o;
    #pragma unroll
    for (int e = 0; e < 4; ++e) o.s[e] = f2bf(f.f[e]);
    *(US4*)(dst + (size_t)idx * 4) = o;
}

// ============================================================================
// Kernel 1: pos_embed f32 -> bf16, padded by 64 zero rows (band overshoot).
// ============================================================================
__global__ void prep_pos_kernel(const float* __restrict__ pos,
                                unsigned short* __restrict__ posb)
{
    int idx = blockIdx.x * 256 + threadIdx.x;          // < 2112*64
    float v = (idx < 2048*64) ? pos[idx] : 0.0f;
    posb[idx] = f2bf(v);
}

// ============================================================================
// Kernel 2: qkv = x @ W_qkv + b_qkv from pre-converted bf16 operands.
//   qh = q + r_r_bias (folded), kh = k, vh = v, per-head [b][h][l][d] bf16.
//   Block = 8 waves -> 128(M) x 128(N) tile; 8 WMMA per K-step per wave.
// ============================================================================
__global__ void __launch_bounds__(256) qkv_gemm_kernel(
    const unsigned short* __restrict__ xb, const unsigned short* __restrict__ Wb,
    const float* __restrict__ bq, const float* __restrict__ rrb,
    unsigned short* __restrict__ qh, unsigned short* __restrict__ kh,
    unsigned short* __restrict__ vh)
{
    __shared__ unsigned short Wt[128 * 32];   // [n_local][k_local] bf16 (transposed)

    const int tid = threadIdx.x;
    const int lane = tid & 31, wid = tid >> 5;
    const int lane15 = lane & 15, half = lane >> 4;
    const int m0 = blockIdx.x * 128;
    const int n0 = blockIdx.y * 128;
    const int mrow = m0 + wid * 16 + lane15;

    v8f zero = {0.f,0.f,0.f,0.f,0.f,0.f,0.f,0.f};
    v8f acc[8];
    #pragma unroll
    for (int i = 0; i < 8; ++i) acc[i] = zero;

    for (int kc = 0; kc < 1024; kc += 32) {
        __syncthreads();
        {   // stage W_bf[kc..kc+31][n0..n0+127] transposed into LDS
            int ngrp = tid & 15;              // 16 groups of 8 columns
            int kk0  = tid >> 4;              // 0..15
            #pragma unroll
            for (int r = 0; r < 2; ++r) {
                int kk = kk0 + 16 * r;
                unsigned short t8[8];
                *(U128*)t8 = *(const U128*)(Wb + (size_t)(kc + kk) * 3072 + n0 + ngrp * 8);
                #pragma unroll
                for (int e = 0; e < 8; ++e) Wt[(ngrp*8 + e)*32 + kk] = t8[e];
            }
        }
        __syncthreads();

        V16 a;   // A operand: x row `mrow`, K-chunk kc..kc+31 (contiguous k = 16*half + s)
        a.q[0] = *(const U128*)(xb + (size_t)mrow * 1024 + kc + 16*half);
        a.q[1] = *(const U128*)(xb + (size_t)mrow * 1024 + kc + 16*half + 8);

        #pragma unroll
        for (int nc = 0; nc < 8; ++nc) {
            V16 b;
            const U128* bp = (const U128*)(Wt + (16*nc + lane15)*32 + 16*half);
            b.q[0] = bp[0]; b.q[1] = bp[1];
            acc[nc] = WMMA_BF16(a.v, b.v, acc[nc]);
        }
    }

    // epilogue: + bias (+ r_r_bias for q third), scatter to [b][h][l][d] bf16
    const int m_base = m0 + wid * 16 + 8 * half;
    #pragma unroll
    for (int nc = 0; nc < 8; ++nc) {
        int ngl   = n0 + 16*nc + lane15;
        int which = ngl >> 10;
        int rem   = ngl & 1023;
        int h = rem >> 6, d = rem & 63;
        float bias = bq[ngl];
        float rr   = (which == 0) ? rrb[h*64 + d] : 0.0f;
        unsigned short* dst = (which == 0) ? qh : (which == 1) ? kh : vh;
        #pragma unroll
        for (int g = 0; g < 8; ++g) {
            int m = m_base + g;
            int bb = m >> 10, l = m & 1023;
            dst[((size_t)(bb*16 + h)*1024 + l)*64 + d] = f2bf(acc[nc][g] + bias + rr);
        }
    }
}

// ============================================================================
// Kernel 3: U = q + k + r_w_bias = qh + kh + (rwb - rrb)   (bf16)
// ============================================================================
__global__ void prep_u_kernel(const unsigned short* __restrict__ qh,
                              const unsigned short* __restrict__ kh,
                              const float* __restrict__ rrb,
                              const float* __restrict__ rwb,
                              unsigned short* __restrict__ U)
{
    int idx = blockIdx.x * 256 + threadIdx.x;          // < 4*16*1024*64
    int d = idx & 63;
    int h = (idx >> 16) & 15;
    float v = bf2f(qh[idx]) + bf2f(kh[idx]) + (rwb[h*64 + d] - rrb[h*64 + d]);
    U[idx] = f2bf(v);
}

// ============================================================================
// Kernel 4: fused relative-position flash attention.
//   grid = (L/128, B*NH); block = 8 waves; wave owns 16 query rows.
//   K tile staged to LDS via TDM (tensor_load_to_lds) when available.
//   Per 32-key chunk: 4 WMMA (q@k^T) + 6 WMMA (16x48 band U@pos^T) + 4 WMMA (P@V).
//   Score[i][j] = (AC[i][j] + Band[i][j-i+15]) * (1/8), online softmax.
// ============================================================================
__global__ void __launch_bounds__(256) rel_attn_kernel(
    const unsigned short* __restrict__ qh, const unsigned short* __restrict__ kh,
    const unsigned short* __restrict__ vh, const unsigned short* __restrict__ U,
    const unsigned short* __restrict__ posb, float* __restrict__ out)
{
    __shared__ unsigned short Klds[32 * 64];    // K tile   [k][d]  (bf16, row-major)
    __shared__ unsigned short Vt[64 * 32];      // V^T tile [d][k]  (bf16)
    __shared__ float          Gb[8][16 * 48];   // per-wave band scratch
    __shared__ unsigned short Pl[8][16 * 32];   // per-wave P (row-major bf16)

    const int tid = threadIdx.x;
    const int lane = tid & 31, wid = tid >> 5;
    const int lane15 = lane & 15, half = lane >> 4;
    const int bb = blockIdx.y >> 4, h = blockIdx.y & 15;
    const int q0 = blockIdx.x * 128 + wid * 16;

    const size_t headoff = (size_t)(bb*16 + h) * 1024 * 64;
    const unsigned short* qp = qh + headoff;
    const unsigned short* kp = kh + headoff;
    const unsigned short* vp = vh + headoff;
    const unsigned short* up = U  + headoff;

    // q and U A-operands, resident for the whole key loop (2 K-chunks d0=0,32)
    V16 qA[2], uA[2];
    #pragma unroll
    for (int c = 0; c < 2; ++c) {
        size_t base = (size_t)(q0 + lane15) * 64 + 32*c;
        qA[c].q[0] = *(const U128*)(qp + base + 16*half);
        qA[c].q[1] = *(const U128*)(qp + base + 16*half + 8);
        uA[c].q[0] = *(const U128*)(up + base + 16*half);
        uA[c].q[1] = *(const U128*)(up + base + 16*half + 8);
    }

    float m_run[8], l_run[8];
    #pragma unroll
    for (int g = 0; g < 8; ++g) { m_run[g] = -1e30f; l_run[g] = 0.0f; }
    v8f zero = {0.f,0.f,0.f,0.f,0.f,0.f,0.f,0.f};
    v8f O[4];
    #pragma unroll
    for (int t = 0; t < 4; ++t) O[t] = zero;

    for (int kc0 = 0; kc0 < 1024; kc0 += 32) {
        __syncthreads();
#if USE_TDM
        if (wid == 0) {   // async DMA of 32x64 bf16 K tile into LDS (wave 0 issues)
            unsigned long long ga = (unsigned long long)(size_t)(kp + (size_t)kc0 * 64);
            unsigned lds_dst = (unsigned)(size_t)(void*)Klds;
            u32x4 g0;
            g0[0] = 1u;                                           // count=1, user desc
            g0[1] = lds_dst;                                      // LDS byte address
            g0[2] = (unsigned)ga;                                 // global_addr lo
            g0[3] = ((unsigned)(ga >> 32) & 0x01FFFFFFu) | (2u << 30); // addr hi | type=2
            i32x8 g1;
            g1[0] = (int)(1u << 16);                              // data_size = 2B
            g1[1] = (int)(64u << 16);                             // tensor_dim0 = 64 (lo16 @bit48)
            g1[2] = (int)(1024u << 16);                           // dim0 hi=0 | tensor_dim1=1024 lo
            g1[3] = (int)(64u << 16);                             // dim1 hi=0 | tile_dim0 = 64
            g1[4] = (int)32u;                                     // tile_dim1 = 32, tile_dim2 = 0
            g1[5] = (int)64u;                                     // tensor_dim0_stride = 64
            g1[6] = 0; g1[7] = 0;
            i32x4 gz = {0, 0, 0, 0};
            __builtin_amdgcn_tensor_load_to_lds(g0, g1, gz, gz, 0);
        }
#else
        {   // fallback: cooperative straight copy of K tile (row-major)
            int key = tid >> 3, dg = tid & 7;
            *(U128*)&Klds[key*64 + dg*8] =
                *(const U128*)(kp + (size_t)(kc0 + key)*64 + dg*8);
        }
#endif
        {   // stage V chunk transposed: Vt[d][k]   (overlaps with TDM)
            int key = tid >> 3, dg = tid & 7;
            U128 raw = *(const U128*)(vp + (size_t)(kc0 + key)*64 + dg*8);
            unsigned short t8[8];
            *(U128*)t8 = raw;
            #pragma unroll
            for (int e = 0; e < 8; ++e) Vt[(dg*8 + e)*32 + key] = t8[e];
        }
#if USE_TDM
        if (wid == 0) __builtin_amdgcn_s_wait_tensorcnt(0);
#endif
        __syncthreads();
        if (kc0 + 32 < 1024)
            __builtin_prefetch(kp + (size_t)(kc0 + 32 + lane)*64, 0, 1);

        // ---- S = q̂ @ k^T  (two 16-key column tiles, K=64 as 2x32) ----
        v8f S[2]; S[0] = zero; S[1] = zero;
        #pragma unroll
        for (int jc = 0; jc < 2; ++jc) {
            #pragma unroll
            for (int c = 0; c < 2; ++c) {
                V16 b;
                const U128* bp = (const U128*)(Klds + (16*jc + lane15)*64 + 32*c + 16*half);
                b.q[0] = bp[0]; b.q[1] = bp[1];
                S[jc] = WMMA_BF16(qA[c].v, b.v, S[jc]);
            }
        }

        // ---- band: G(16x48) = U @ pos[rbase..rbase+47]^T ----
        const int rbase = kc0 - q0 + 1024 - 15;   // >= 1; rbase+47 <= 2048 (padded)
        v8f G[3]; G[0] = zero; G[1] = zero; G[2] = zero;
        #pragma unroll
        for (int nc = 0; nc < 3; ++nc) {
            #pragma unroll
            for (int c = 0; c < 2; ++c) {
                V16 b;
                const U128* bp = (const U128*)(posb + (size_t)(rbase + 16*nc + lane15)*64 + 32*c + 16*half);
                b.q[0] = bp[0]; b.q[1] = bp[1];
                G[nc] = WMMA_BF16(uA[c].v, b.v, G[nc]);
            }
        }
        #pragma unroll
        for (int nc = 0; nc < 3; ++nc)
            #pragma unroll
            for (int g = 0; g < 8; ++g)
                Gb[wid][(g + 8*half)*48 + 16*nc + lane15] = G[nc][g];

        // ---- assemble scores, online softmax (rows live in 16-lane halves) ----
        float p0[8], p1[8];
        #pragma unroll
        for (int g = 0; g < 8; ++g) {
            int i = g + 8*half;
            float s0 = (S[0][g] + Gb[wid][i*48 + (lane15      - i + 15)]) * 0.125f;
            float s1 = (S[1][g] + Gb[wid][i*48 + (lane15 + 16 - i + 15)]) * 0.125f;
            float mx = fmaxf(s0, s1);
            #pragma unroll
            for (int off = 1; off < 16; off <<= 1) mx = fmaxf(mx, __shfl_xor(mx, off, 32));
            float mnew  = fmaxf(m_run[g], mx);
            float alpha = __expf(m_run[g] - mnew);
            float e0 = __expf(s0 - mnew), e1 = __expf(s1 - mnew);
            float rs = e0 + e1;
            #pragma unroll
            for (int off = 1; off < 16; off <<= 1) rs += __shfl_xor(rs, off, 32);
            l_run[g] = l_run[g] * alpha + rs;
            m_run[g] = mnew;
            p0[g] = e0; p1[g] = e1;
            #pragma unroll
            for (int t = 0; t < 4; ++t) O[t][g] *= alpha;
        }

        // ---- P: D-layout -> LDS row-major -> A-layout ----
        #pragma unroll
        for (int g = 0; g < 8; ++g) {
            int i = g + 8*half;
            Pl[wid][i*32 + lane15]      = f2bf(p0[g]);
            Pl[wid][i*32 + 16 + lane15] = f2bf(p1[g]);
        }
        V16 pA;
        pA.q[0] = *(const U128*)(&Pl[wid][lane15*32 + 16*half]);
        pA.q[1] = *(const U128*)(&Pl[wid][lane15*32 + 16*half + 8]);

        // ---- O += P @ V  (4 d-chunks, K=32 keys) ----
        #pragma unroll
        for (int t = 0; t < 4; ++t) {
            V16 b;
            const U128* bp = (const U128*)(Vt + (16*t + lane15)*32 + 16*half);
            b.q[0] = bp[0]; b.q[1] = bp[1];
            O[t] = WMMA_BF16(pA.v, b.v, O[t]);
        }
    }

    // epilogue: out[b, l, h*64 + d] = O / l_run
    #pragma unroll
    for (int t = 0; t < 4; ++t) {
        #pragma unroll
        for (int g = 0; g < 8; ++g) {
            int i = g + 8*half;
            int row = q0 + i;
            int d = 16*t + lane15;
            out[((size_t)bb*1024 + row)*1024 + h*64 + d] = O[t][g] / l_run[g];
        }
    }
}

// ============================================================================
extern "C" void kernel_launch(void* const* d_in, const int* in_sizes, int n_in,
                              void* d_out, int out_size, void* d_ws, size_t ws_size,
                              hipStream_t stream)
{
    const float* x   = (const float*)d_in[0];
    const float* pos = (const float*)d_in[1];
    const float* W   = (const float*)d_in[2];
    const float* bq  = (const float*)d_in[3];
    const float* rrb = (const float*)d_in[4];
    const float* rwb = (const float*)d_in[5];
    float* out = (float*)d_out;

    char* ws = (char*)d_ws;
    unsigned short* qh   = (unsigned short*)(ws);                        // 8 MB
    unsigned short* kh   = (unsigned short*)(ws + ((size_t)8  << 20));   // 8 MB
    unsigned short* vh   = (unsigned short*)(ws + ((size_t)16 << 20));   // 8 MB
    unsigned short* U    = (unsigned short*)(ws + ((size_t)24 << 20));   // 8 MB
    unsigned short* posb = (unsigned short*)(ws + ((size_t)32 << 20));   // 264 KB
    unsigned short* xb   = (unsigned short*)(ws + ((size_t)33 << 20));   // 8 MB
    unsigned short* Wb   = (unsigned short*)(ws + ((size_t)41 << 20));   // 6 MB

    cvt_bf16_kernel<<<dim3(4096), dim3(256), 0, stream>>>(x, xb, 4096*1024/4);
    cvt_bf16_kernel<<<dim3(3072), dim3(256), 0, stream>>>(W, Wb, 1024*3072/4);
    prep_pos_kernel<<<dim3((2112*64)/256), dim3(256), 0, stream>>>(pos, posb);
    qkv_gemm_kernel<<<dim3(32, 24), dim3(256), 0, stream>>>(xb, Wb, bq, rrb, qh, kh, vh);
    prep_u_kernel<<<dim3((4*16*1024*64)/256), dim3(256), 0, stream>>>(qh, kh, rrb, rwb, U);
    rel_attn_kernel<<<dim3(8, 64), dim3(256), 0, stream>>>(qh, kh, vh, U, posb, out);
}
